// SWMSA_67748814127319
// MI455X (gfx1250) — matrix operand order repaired and checked by reading, not compiled, not run on hardware
//
#include <hip/hip_runtime.h>
#include <hip/hip_bf16.h>

typedef __attribute__((ext_vector_type(16))) _Float16 v16h;
typedef __attribute__((ext_vector_type(8)))  float    v8f;

#define WSZ   7
#define SSZ   3
#define NH    3
#define HD    32
#define NTOK  49
#define NPAD  64
#define CCH   96
#define C3    288
#define HH    112

// padded LDS row strides (halves): multiples of 8 halves (16B) for b128 frags,
// byte strides chosen to avoid LDS bank conflicts
#define LDX   104   // Xs / Ot / Wq / Wp row stride
#define LDQ   200   // QKV (Q|K) row stride
#define LDP   72    // P row stride
#define LDS_S 68    // S row stride (floats)
#define LDV   72    // Vt row stride

// ---- LDS layout (bytes), phase-overlapped ----
#define OFF_XS     0          // phase1: X window   64 x 104 f16  (13312)
#define OFF_WQKV   13312      // phase1: qkv_w^T  [288][104] f16  (59904) -> 73216
#define OFF_S      0          // phase2: scores     64 x 68  f32  (17408)
#define OFF_P      17408      // phase2: softmax    64 x 72  f16  ( 9216) -> 26624
#define OFF_OUT    26624      // phase2: attn out   64 x 104 f16  (13312) -> 39936
#define OFF_WPROJ  39936      // phase3: proj_w^T  [96][104] f16  (19968) -> 59904
#define OFF_BIAS   59904      // bias_table copy   169x3     f32  ( 2048) -> 61952
#define OFF_IDX    61952      // bias/mask idx     64x64     u16  ( 8192) -> 70144 (< 73216)
#define OFF_QKV    73216      // Q|K cols 0..191   64 x 200  f16  (25600) -> 98816
#define OFF_VT     98816      // V^T [96][72]                f16  (13824) -> 112640
#define LDS_BYTES  112640

#define WMMA_F16(A, B, C) \
  __builtin_amdgcn_wmma_f32_16x16x32_f16(false, (A), false, (B), (short)0, (C), false, false)

// A fragment 16x32 f16 (ISA 7.12.2): lanes 0-15: M=lane, K {0-7,16-23};
// lanes 16-31: M=lane-16, K {8-15,24-31}.  -> two ds_load_b128 per lane.
__device__ __forceinline__ v16h load_a_frag(const _Float16* base, int ld, int row0, int k0, int lane) {
  const int m  = lane & 15;
  const int kh = (lane >> 4) << 3;
  const _Float16* p = base + (row0 + m) * ld + k0 + kh;
  v16h a;
#pragma unroll
  for (int e = 0; e < 16; ++e) {
    int v = e >> 1, hb = e & 1;
    int k = ((v & 4) << 2) + ((v & 3) << 1) + hb;
    a[e] = p[k];
  }
  return a;
}

// B fragment 32x16 f16: lanes 0-15 = col n, K {0-7,16-23}; lanes 16-31 = col n, K {8-15,24-31}.
__device__ __forceinline__ v16h load_b_frag(const _Float16* base, int kstride, int nstride,
                                            int k0, int n0, int lane) {
  const int n  = lane & 15;
  const int kb = (lane >> 4) << 3;
  const _Float16* p = base + (k0 + kb) * kstride + (n0 + n) * nstride;
  v16h b;
#pragma unroll
  for (int e = 0; e < 16; ++e) b[e] = p[(e + ((e >> 3) << 3)) * kstride];
  return b;
}

__device__ __forceinline__ int region_cat(int u) { return u < 105 ? 0 : (u < 109 ? 1 : 2); }

__global__ __launch_bounds__(256) void swmsa_fused_kernel(
    const float* __restrict__ x, const float* __restrict__ qkv_w,
    const float* __restrict__ qkv_b, const float* __restrict__ proj_w,
    const float* __restrict__ proj_b, const float* __restrict__ bias_table,
    float* __restrict__ out)
{
  extern __shared__ char smem[];
  _Float16* Xs  = (_Float16*)(smem + OFF_XS);
  _Float16* Wq  = (_Float16*)(smem + OFF_WQKV);
  float*    S   = (float*)   (smem + OFF_S);
  _Float16* P   = (_Float16*)(smem + OFF_P);
  _Float16* Ot  = (_Float16*)(smem + OFF_OUT);
  _Float16* Wp  = (_Float16*)(smem + OFF_WPROJ);
  float*    Bt  = (float*)   (smem + OFF_BIAS);
  unsigned short* IDXT = (unsigned short*)(smem + OFF_IDX);
  _Float16* QKV = (_Float16*)(smem + OFF_QKV);
  _Float16* Vt  = (_Float16*)(smem + OFF_VT);

  const int tid  = threadIdx.x;
  const int wave = tid >> 5;
  const int lane = tid & 31;
  const int w  = blockIdx.x;           // window id: b*256 + wi*16 + wj
  const int bb = w >> 8;
  const int wi = (w >> 4) & 15;
  const int wj = w & 15;
  const float scale = 0.17677669529663687f;   // 32^-0.5
  const int tm = wave & 3;             // per-wave row-tile (fixed for all GEMMs)
  const int tn0 = wave >> 2;           // per-wave starting col-tile (step 2)

  // ---- stage shifted X window -> LDS f16, zero-pad rows 49..63 ----
  for (int e = tid; e < NPAD * CCH; e += 256) {
    int p = e / CCH, ch = e - p * CCH;
    float v = 0.f;
    if (p < NTOK) {
      int r = p / WSZ, c = p - r * WSZ;
      int hh = wi * WSZ + r + SSZ; if (hh >= HH) hh -= HH;   // roll(-ss)
      int ww = wj * WSZ + c + SSZ; if (ww >= HH) ww -= HH;
      v = x[((bb * HH + hh) * HH + ww) * CCH + ch];
    }
    Xs[p * LDX + ch] = (_Float16)v;
  }
  // ---- stage qkv_w transposed [n][k] ----
  for (int e = tid; e < CCH * C3; e += 256) {
    int k = e / C3, n = e - k * C3;
    Wq[n * LDX + k] = (_Float16)qkv_w[e];
  }
  __syncthreads();

  // ================= Phase 1: QKV = Xs @ W + b (q scaled) =================
  {
    // A fragments are invariant across this wave's 9 tiles: load once
    v16h a0 = load_a_frag(Xs, LDX, tm * 16,  0, lane);
    v16h a1 = load_a_frag(Xs, LDX, tm * 16, 32, lane);
    v16h a2 = load_a_frag(Xs, LDX, tm * 16, 64, lane);
    for (int tn = tn0; tn < 18; tn += 2) {
      v16h b0 = load_b_frag(Wq, 1, LDX,  0, tn * 16, lane);
      v16h b1 = load_b_frag(Wq, 1, LDX, 32, tn * 16, lane);
      v16h b2 = load_b_frag(Wq, 1, LDX, 64, tn * 16, lane);
      v8f acc = {};
      acc = WMMA_F16(a0, b0, acc);
      acc = WMMA_F16(a1, b1, acc);
      acc = WMMA_F16(a2, b2, acc);
      int n = tn * 16 + (lane & 15);
      float bqn  = qkv_b[n];
      float mult = (n < CCH) ? scale : 1.f;     // q * hd^-0.5
      // branchless dual-layout store: Q|K row-major, V transposed
      int off0 = OFF_QKV + n * 2;                 // + m*2*LDQ
      int off1 = OFF_VT + (n - 192) * (2 * LDV);  // + m*2
      bool toQK = (n < 192);
#pragma unroll
      for (int j = 0; j < 8; ++j) {
        int m = tm * 16 + ((lane >> 4) << 3) + j;
        float val = (acc[j] + bqn) * mult;
        int off = toQK ? (off0 + m * (2 * LDQ)) : (off1 + m * 2);
        *(_Float16*)(smem + off) = (_Float16)val;
      }
    }
  }
  __syncthreads();

  // ---- stage proj_w^T, bias_table, and combined bias/mask index table ----
  for (int e = tid; e < CCH * CCH; e += 256) {
    int k = e / CCH, n = e - k * CCH;
    Wp[n * LDX + k] = (_Float16)proj_w[e];
  }
  for (int e = tid; e < 169 * NH; e += 256) Bt[e] = bias_table[e];
  for (int e = tid; e < NPAD * NPAD; e += 256) {
    int m = e >> 6, n = e & 63;
    unsigned short u = 0xFFFFu;               // padded or masked -> -1e9
    if (m < NTOK && n < NTOK) {
      int ri = m / WSZ, ci = m - ri * WSZ;
      int rj = n / WSZ, cj = n - rj * WSZ;
      int hi = wi * WSZ + ri + SSZ; if (hi >= HH) hi -= HH;
      int wa = wj * WSZ + ci + SSZ; if (wa >= HH) wa -= HH;
      int hj = wi * WSZ + rj + SSZ; if (hj >= HH) hj -= HH;
      int wb = wj * WSZ + cj + SSZ; if (wb >= HH) wb -= HH;
      int regi = region_cat(hi) * 3 + region_cat(wa);
      int regj = region_cat(hj) * 3 + region_cat(wb);
      if (regi == regj) u = (unsigned short)((ri - rj + 6) * 13 + (ci - cj + 6));
    }
    IDXT[e] = u;
  }
  __syncthreads();

  // ================= Phase 2: per-head attention =================
  for (int h = 0; h < NH; ++h) {
    // S = Qh @ Kh^T : 4x4 tiles, 2 per wave sharing one A (Q) fragment
    {
      v16h a = load_a_frag(QKV, LDQ, tm * 16, h * HD, lane);
      v16h bA = load_b_frag(QKV + CCH + h * HD, 1, LDQ, 0, tn0 * 16, lane);
      v16h bB = load_b_frag(QKV + CCH + h * HD, 1, LDQ, 0, (tn0 + 2) * 16, lane);
#pragma unroll
      for (int half = 0; half < 2; ++half) {
        int tn = tn0 + half * 2;
        v8f acc = {};
        acc = WMMA_F16(a, half ? bB : bA, acc);
        int n = tn * 16 + (lane & 15);
#pragma unroll
        for (int j = 0; j < 8; ++j) {
          int m = tm * 16 + ((lane >> 4) << 3) + j;
          unsigned short u = IDXT[(m << 6) + n];
          float val = (u == 0xFFFFu) ? -1e9f : (acc[j] + Bt[u * NH + h]);
          S[m * LDS_S + n] = val;
        }
      }
    }
    __syncthreads();

    // row softmax: 4 lanes per row (contiguous quads within a wave)
    {
      int row = tid >> 2, q = tid & 3;
      const float* srow = S + row * LDS_S + q * 16;
      float lm = -1e30f;
#pragma unroll
      for (int i = 0; i < 16; ++i) lm = fmaxf(lm, srow[i]);
      lm = fmaxf(lm, __shfl_xor(lm, 1, 32));
      lm = fmaxf(lm, __shfl_xor(lm, 2, 32));
      float ev[16], ls = 0.f;
#pragma unroll
      for (int i = 0; i < 16; ++i) { ev[i] = __expf(srow[i] - lm); ls += ev[i]; }
      ls += __shfl_xor(ls, 1, 32);
      ls += __shfl_xor(ls, 2, 32);
      float inv = 1.f / ls;
      _Float16* prow = P + row * LDP + q * 16;
#pragma unroll
      for (int i = 0; i < 16; ++i) prow[i] = (_Float16)(ev[i] * inv);
    }
    __syncthreads();

    // O = P @ Vh : 4x2 tiles (one per wave), 2 K-chunks, all frags preloaded
    {
      int tnv = wave >> 2;               // 0..1
      v16h a0 = load_a_frag(P, LDP, tm * 16,  0, lane);
      v16h a1 = load_a_frag(P, LDP, tm * 16, 32, lane);
      v16h b0 = load_b_frag(Vt + (h * HD) * LDV, 1, LDV,  0, tnv * 16, lane);
      v16h b1 = load_b_frag(Vt + (h * HD) * LDV, 1, LDV, 32, tnv * 16, lane);
      v8f acc = {};
      acc = WMMA_F16(a0, b0, acc);
      acc = WMMA_F16(a1, b1, acc);
      int n = h * HD + tnv * 16 + (lane & 15);
#pragma unroll
      for (int j = 0; j < 8; ++j) {
        int m = tm * 16 + ((lane >> 4) << 3) + j;
        Ot[m * LDX + n] = (_Float16)acc[j];
      }
    }
    __syncthreads();
  }

  // ================= Phase 3: Y = Ot @ proj_w + b, reverse shift/store ====
  {
    v16h a0 = load_a_frag(Ot, LDX, tm * 16,  0, lane);
    v16h a1 = load_a_frag(Ot, LDX, tm * 16, 32, lane);
    v16h a2 = load_a_frag(Ot, LDX, tm * 16, 64, lane);
    for (int tn = tn0; tn < 6; tn += 2) {
      v16h b0 = load_b_frag(Wp, 1, LDX,  0, tn * 16, lane);
      v16h b1 = load_b_frag(Wp, 1, LDX, 32, tn * 16, lane);
      v16h b2 = load_b_frag(Wp, 1, LDX, 64, tn * 16, lane);
      v8f acc = {};
      acc = WMMA_F16(a0, b0, acc);
      acc = WMMA_F16(a1, b1, acc);
      acc = WMMA_F16(a2, b2, acc);
      int n = tn * 16 + (lane & 15);
      float pb = proj_b[n];
#pragma unroll
      for (int j = 0; j < 8; ++j) {
        int m = tm * 16 + ((lane >> 4) << 3) + j;
        if (m < NTOK) {
          int r = m / WSZ, c = m - r * WSZ;
          // window-reverse + roll(+ss) == same map as the shifted load
          int hh = wi * WSZ + r + SSZ; if (hh >= HH) hh -= HH;
          int ww = wj * WSZ + c + SSZ; if (ww >= HH) ww -= HH;
          out[((bb * HH + hh) * HH + ww) * CCH + n] = acc[j] + pb;
        }
      }
    }
  }
}

extern "C" void kernel_launch(void* const* d_in, const int* in_sizes, int n_in,
                              void* d_out, int out_size, void* d_ws, size_t ws_size,
                              hipStream_t stream) {
  (void)in_sizes; (void)n_in; (void)out_size; (void)d_ws; (void)ws_size;
  const float* x     = (const float*)d_in[0];
  const float* qkv_w = (const float*)d_in[1];
  const float* qkv_b = (const float*)d_in[2];
  const float* pw    = (const float*)d_in[3];
  const float* pb    = (const float*)d_in[4];
  const float* bt    = (const float*)d_in[5];
  float* y = (float*)d_out;

  hipFuncSetAttribute((const void*)swmsa_fused_kernel,
                      hipFuncAttributeMaxDynamicSharedMemorySize, LDS_BYTES);

  swmsa_fused_kernel<<<dim3(8192), dim3(256), LDS_BYTES, stream>>>(
      x, qkv_w, qkv_b, pw, pb, bt, y);
}
